// BertModelToken_4793183502492
// MI455X (gfx1250) — matrix operand order repaired and checked by reading, not compiled, not run on hardware
//
#include <hip/hip_runtime.h>
#include <hip/hip_bf16.h>
#include <math.h>

#define H     256
#define IDIM  512
#define SEQ   64
#define NHEADS 4
#define LAYERS 2
#define NSEQ  1600            // B*V
#define NTOK  (NSEQ * SEQ)    // 102400

typedef __attribute__((ext_vector_type(16))) __bf16        v16bf;
typedef __attribute__((ext_vector_type(16))) unsigned short v16u;
typedef __attribute__((ext_vector_type(8)))  float          v8f;

static __device__ __forceinline__ unsigned short f2bf(float f) {
  unsigned int u = __float_as_uint(f);
  u += 0x7FFFu + ((u >> 16) & 1u);          // round-to-nearest-even
  return (unsigned short)(u >> 16);
}

// ---- CDNA5 async copy: global -> LDS, per-lane 16B, tracked by ASYNCcnt ----
// GVS addressing: mem = SADDR(u64) + VADDR(i32); LDS byte addr in vdst VGPR.
static __device__ __forceinline__ void async_b128(unsigned lds_addr, const void* base,
                                                  unsigned voff) {
  asm volatile("global_load_async_to_lds_b128 %0, %1, %2"
               :: "v"(lds_addr), "v"(voff), "s"(base) : "memory");
}
static __device__ __forceinline__ void wait_async0() {
  asm volatile("s_wait_asynccnt 0x0" ::: "memory");
}
static __device__ __forceinline__ unsigned lds_addr_of(const void* p) {
  return (unsigned)(uintptr_t)p;   // LDS aperture: addr[31:0] is the LDS byte address
}

// ---- WMMA fragment loaders (bf16 16x16x32, wave32 layouts per CDNA5 ISA) ----
// A (16x32, MxK): lane 0-15 -> M=lane, K={0..7,16..23}; lane 16-31 -> M=lane-16, K={8..15,24..31}
static __device__ __forceinline__ v16bf load_a_frag(const unsigned short* base, int stride,
                                                    int lane, int kofs) {
  const int row = lane & 15;
  const int kb  = (lane >> 4) * 8;
  const unsigned short* p0 = base + row * stride + kofs + kb;
  const unsigned short* p1 = base + row * stride + kofs + 16 + kb;
  v16u v;
#pragma unroll
  for (int i = 0; i < 8; ++i) { v[i] = p0[i]; v[8 + i] = p1[i]; }
  return __builtin_bit_cast(v16bf, v);
}
// B (32x16, KxN): lane 0-15 -> N=lane, K={0..7,16..23}; lane 16-31 -> N=lane-16, K={8..15,24..31}
// base points at an N-major (transposed) tile: base[n*stride + k]
static __device__ __forceinline__ v16bf load_b_frag(const unsigned short* base, int stride,
                                                    int lane, int kofs) {
  const int col = lane & 15;
  const int kb  = (lane >> 4) * 8;
  const unsigned short* p = base + col * stride + kofs + kb;
  v16u v;
#pragma unroll
  for (int i = 0; i < 8; ++i) { v[i] = p[i]; v[8 + i] = p[16 + i]; }
  return __builtin_bit_cast(v16bf, v);
}

// ------------------------- generic tiled WMMA GEMM --------------------------
// C[M, Nout] = A(bf16)[M,K] @ W + bias; W is PRE-TRANSPOSED: Wt[Nout][K] so both
// LDS tiles are straight async copies. Block: 256 thr = 8 waves, 64x128 tile,
// double-buffered async global->LDS staging.
// wave w -> msub = w&3, nsub = (w>>2)*4 + {0..3}  (32 subtiles of 16x16)
enum { EPI_BF16 = 0, EPI_RES_F32 = 1, EPI_GELU_BF16 = 2, EPI_TANH_F32 = 3 };

template <int EPI>
__global__ __launch_bounds__(256)
void gemm_wmma(const unsigned short* __restrict__ A, int lda,
               const unsigned short* __restrict__ Wt,   // [Nout][K] bf16
               const float* __restrict__ bias,
               const float* __restrict__ res,
               float* __restrict__ outf,
               unsigned short* __restrict__ outb,
               int K) {
  __shared__ unsigned short As[2][64 * 32];    // [buf][row][k]
  __shared__ unsigned short Bs[2][128 * 32];   // [buf][ncol][k]  (N-major)

  const int tid  = threadIdx.x;
  const int wave = tid >> 5, lane = tid & 31;
  const int m0 = blockIdx.y * 64, n0 = blockIdx.x * 128;
  const int Nout = gridDim.x * 128;
  const int msub = wave & 3, nsb = (wave >> 2) * 4;

  v8f acc[4] = {{}, {}, {}, {}};

  const int arow = tid >> 2, akg = (tid & 3) * 8;   // A tile: 64x32, 16B/thread
  const int brow = tid >> 1, bkg = (tid & 1) * 16;  // B tile: 128x32, 32B/thread

  const unsigned as_base = lds_addr_of(&As[0][0]);
  const unsigned bs_base = lds_addr_of(&Bs[0][0]);

  auto issue = [&](int buf, int kt) {
    async_b128(as_base + (unsigned)((buf * 64 * 32 + arow * 32 + akg) * 2),
               A, (unsigned)(((m0 + arow) * lda + kt + akg) * 2));
    async_b128(bs_base + (unsigned)((buf * 128 * 32 + brow * 32 + bkg) * 2),
               Wt, (unsigned)(((n0 + brow) * K + kt + bkg) * 2));
    async_b128(bs_base + (unsigned)((buf * 128 * 32 + brow * 32 + bkg + 8) * 2),
               Wt, (unsigned)(((n0 + brow) * K + kt + bkg + 8) * 2));
  };

  const int nk = K >> 5;
  issue(0, 0);
  wait_async0();
  __syncthreads();

  for (int i = 0; i < nk; ++i) {
    const int cur = i & 1;
    if (i + 1 < nk) issue(cur ^ 1, (i + 1) * 32);   // prefetch next tile (async)

    v16bf a = load_a_frag(&As[cur][(msub * 16) * 32], 32, lane, 0);
#pragma unroll
    for (int j = 0; j < 4; ++j) {
      v16bf b = load_b_frag(&Bs[cur][((nsb + j) * 16) * 32], 32, lane, 0);
      acc[j] = __builtin_amdgcn_wmma_f32_16x16x32_bf16(false, a, false, b,
                                                       (short)0, acc[j], false, false);
    }
    if (i + 1 < nk) wait_async0();
    __syncthreads();
  }

  const int rbase = msub * 16 + 8 * (lane >> 4);
  const int cl = lane & 15;
#pragma unroll
  for (int j = 0; j < 4; ++j) {
    const int ncol = n0 + (nsb + j) * 16 + cl;
    const float bv = bias[ncol];
#pragma unroll
    for (int r = 0; r < 8; ++r) {
      const int grow = m0 + rbase + r;
      const size_t oidx = (size_t)grow * Nout + ncol;
      float val = acc[j][r] + bv;
      if (EPI == EPI_BF16) {
        outb[oidx] = f2bf(val);
      } else if (EPI == EPI_RES_F32) {
        outf[oidx] = val + res[oidx];
      } else if (EPI == EPI_GELU_BF16) {
        float g = 0.5f * val * (1.0f + erff(val * 0.70710678118654752f));
        outb[oidx] = f2bf(g);
      } else { // EPI_TANH_F32
        outf[oidx] = tanhf(val);
      }
    }
  }
}

// ---------------- fused attention: one block per (sequence, head) -----------
__global__ __launch_bounds__(256)
void attn_kernel(const unsigned short* __restrict__ Qg,
                 const unsigned short* __restrict__ Kg,
                 const unsigned short* __restrict__ Vg,
                 const float* __restrict__ mask,
                 unsigned short* __restrict__ ctx) {
  __shared__ unsigned short Qs[64 * 64];   // [q][d]
  __shared__ unsigned short Ks[64 * 64];   // [k][d]  (N-major B for Q@K^T)
  __shared__ unsigned short Vt[64 * 64];   // [d][k]  (N-major B for P@V)
  __shared__ float          Sc[64 * 64];   // f32 scores
  __shared__ unsigned short Ps[64 * 64];   // bf16 probabilities
  __shared__ float          am[64];

  const int n = blockIdx.x, h = blockIdx.y;
  const int tid = threadIdx.x, wave = tid >> 5, lane = tid & 31;
  const int tokbase = n * SEQ;
  const int cbase = h * 64;

  const unsigned qs_base = lds_addr_of(Qs);
  const unsigned ks_base = lds_addr_of(Ks);

  for (int idx = tid; idx < 64 * 64 / 8; idx += 256) {
    const int r = idx >> 3, c8 = (idx & 7) * 8;
    const unsigned go = (unsigned)(((tokbase + r) * H + cbase + c8) * 2);
    const unsigned lo = (unsigned)((r * 64 + c8) * 2);
    async_b128(qs_base + lo, Qg, go);     // direct copy
    async_b128(ks_base + lo, Kg, go);     // direct copy
    uint4 vv = *(const uint4*)&Vg[(size_t)(tokbase + r) * H + cbase + c8];
    unsigned short vr[8];
    __builtin_memcpy(vr, &vv, 16);
#pragma unroll
    for (int j = 0; j < 8; ++j) Vt[(c8 + j) * 64 + r] = vr[j];   // transpose
  }
  if (tid < 64) am[tid] = (1.0f - mask[tokbase + tid]) * -10000.0f;
  wait_async0();
  __syncthreads();

  const int qsub = wave & 3;
  const int sub0 = (wave >> 2) * 2;

  // scores = Q @ K^T  (d = 64 contraction -> two k-steps of 32)
#pragma unroll
  for (int t = 0; t < 2; ++t) {
    const int ks = sub0 + t;
    v8f c = {};
#pragma unroll
    for (int kc = 0; kc < 64; kc += 32) {
      v16bf a = load_a_frag(&Qs[(qsub * 16) * 64], 64, lane, kc);
      v16bf b = load_b_frag(&Ks[(ks * 16) * 64], 64, lane, kc);
      c = __builtin_amdgcn_wmma_f32_16x16x32_bf16(false, a, false, b, (short)0, c, false, false);
    }
#pragma unroll
    for (int r = 0; r < 8; ++r)
      Sc[(qsub * 16 + r + 8 * (lane >> 4)) * 64 + ks * 16 + (lane & 15)] = c[r];
  }
  __syncthreads();

  // softmax over each row of 64 (scale + additive mask)
  if (tid < 64) {
    const float scale = 0.125f;  // 1/sqrt(64)
    float* srow = &Sc[tid * 64];
    float mx = -3.0e38f;
#pragma unroll 8
    for (int k = 0; k < 64; ++k) { float s = srow[k] * scale + am[k]; srow[k] = s; mx = fmaxf(mx, s); }
    float sum = 0.0f;
#pragma unroll 8
    for (int k = 0; k < 64; ++k) { float e = __expf(srow[k] - mx); srow[k] = e; sum += e; }
    const float inv = 1.0f / sum;
#pragma unroll 8
    for (int k = 0; k < 64; ++k) Ps[tid * 64 + k] = f2bf(srow[k] * inv);
  }
  __syncthreads();

  // ctx = P @ V
#pragma unroll
  for (int t = 0; t < 2; ++t) {
    const int ds = sub0 + t;
    v8f c = {};
#pragma unroll
    for (int kc = 0; kc < 64; kc += 32) {
      v16bf a = load_a_frag(&Ps[(qsub * 16) * 64], 64, lane, kc);
      v16bf b = load_b_frag(&Vt[(ds * 16) * 64], 64, lane, kc);
      c = __builtin_amdgcn_wmma_f32_16x16x32_bf16(false, a, false, b, (short)0, c, false, false);
    }
#pragma unroll
    for (int r = 0; r < 8; ++r) {
      const int row = qsub * 16 + r + 8 * (lane >> 4);
      const int col = ds * 16 + (lane & 15);
      ctx[(size_t)(tokbase + row) * H + cbase + col] = f2bf(c[r]);
    }
  }
}

// ----------------------- LayerNorm over H=256, 1 block/row ------------------
__global__ __launch_bounds__(256)
void layernorm_kernel(const float* __restrict__ in,
                      const float* __restrict__ g, const float* __restrict__ b,
                      float* __restrict__ outf, unsigned short* __restrict__ outb) {
  __shared__ float red[256];
  const int row = blockIdx.x, tid = threadIdx.x;
  const float v = in[(size_t)row * H + tid];
  red[tid] = v; __syncthreads();
  for (int s = 128; s > 0; s >>= 1) { if (tid < s) red[tid] += red[tid + s]; __syncthreads(); }
  const float mean = red[0] * (1.0f / H);
  __syncthreads();
  const float d = v - mean;
  red[tid] = d * d; __syncthreads();
  for (int s = 128; s > 0; s >>= 1) { if (tid < s) red[tid] += red[tid + s]; __syncthreads(); }
  const float var = red[0] * (1.0f / H);
  const float y = d * rsqrtf(var + 1e-12f) * g[tid] + b[tid];
  outf[(size_t)row * H + tid] = y;
  outb[(size_t)row * H + tid] = f2bf(y);
}

// -------------- embeddings gather + LayerNorm, 1 block per token ------------
__global__ __launch_bounds__(256)
void embed_ln_kernel(const int* __restrict__ word, const int* __restrict__ age,
                     const int* __restrict__ seg, const int* __restrict__ posi,
                     const float* __restrict__ we, const float* __restrict__ se,
                     const float* __restrict__ ae, const float* __restrict__ pe,
                     const float* __restrict__ g, const float* __restrict__ b,
                     float* __restrict__ xf, unsigned short* __restrict__ xb) {
  __shared__ float red[256];
  const int tok = blockIdx.x, tid = threadIdx.x;
  const int w = word[tok], a_ = age[tok], s_ = seg[tok], p = posi[tok];
  const float v = we[(size_t)w * H + tid] + se[(size_t)s_ * H + tid] +
                  ae[(size_t)a_ * H + tid] + pe[(size_t)p * H + tid];
  red[tid] = v; __syncthreads();
  for (int s = 128; s > 0; s >>= 1) { if (tid < s) red[tid] += red[tid + s]; __syncthreads(); }
  const float mean = red[0] * (1.0f / H);
  __syncthreads();
  const float d = v - mean;
  red[tid] = d * d; __syncthreads();
  for (int s = 128; s > 0; s >>= 1) { if (tid < s) red[tid] += red[tid + s]; __syncthreads(); }
  const float var = red[0] * (1.0f / H);
  const float y = d * rsqrtf(var + 1e-12f) * g[tid] + b[tid];
  xf[(size_t)tok * H + tid] = y;
  xb[(size_t)tok * H + tid] = f2bf(y);
}

// ---- weight convert f32[L][K][N] -> bf16 TRANSPOSED [L][N][K] --------------
__global__ __launch_bounds__(256)
void convert_T_kernel(const float* __restrict__ in, unsigned short* __restrict__ out,
                      int K, int N, int total) {
  const int idx = blockIdx.x * 256 + threadIdx.x;
  if (idx >= total) return;
  const int kn = K * N;
  const int l = idx / kn;
  const int r = idx - l * kn;
  const int k = r / N, n2 = r - k * N;
  out[(size_t)l * kn + (size_t)n2 * K + k] = f2bf(in[idx]);
}

// ------------------------------- orchestration ------------------------------
extern "C" void kernel_launch(void* const* d_in, const int* in_sizes, int n_in,
                              void* d_out, int out_size, void* d_ws, size_t ws_size,
                              hipStream_t stream) {
  const int*   word     = (const int*)d_in[0];
  const int*   age      = (const int*)d_in[1];
  const int*   seg      = (const int*)d_in[2];
  const int*   posi     = (const int*)d_in[3];
  const float* mask     = (const float*)d_in[4];
  // d_in[5] = token (unused: pooler takes hidden[:, 0])
  const float* word_emb = (const float*)d_in[6];
  const float* seg_emb  = (const float*)d_in[7];
  const float* age_emb  = (const float*)d_in[8];
  const float* posi_emb = (const float*)d_in[9];
  const float* emb_ln_g = (const float*)d_in[10];
  const float* emb_ln_b = (const float*)d_in[11];
  const float* qw = (const float*)d_in[12];
  const float* qb = (const float*)d_in[13];
  const float* kw = (const float*)d_in[14];
  const float* kb = (const float*)d_in[15];
  const float* vw = (const float*)d_in[16];
  const float* vb = (const float*)d_in[17];
  const float* ow = (const float*)d_in[18];
  const float* ob = (const float*)d_in[19];
  const float* attn_ln_g = (const float*)d_in[20];
  const float* attn_ln_b = (const float*)d_in[21];
  const float* iw = (const float*)d_in[22];
  const float* ib = (const float*)d_in[23];
  const float* outw = (const float*)d_in[24];
  const float* ffn_ob = (const float*)d_in[25];
  const float* out_ln_g = (const float*)d_in[26];
  const float* out_ln_b = (const float*)d_in[27];
  const float* pool_w = (const float*)d_in[28];
  const float* pool_b = (const float*)d_in[29];
  float* out = (float*)d_out;

  char* ws = (char*)d_ws;
  size_t off = 0;
  auto alloc = [&](size_t bytes) -> char* {
    char* p = ws + off;
    off += (bytes + 255) & ~(size_t)255;
    return p;
  };

  float* xf   = (float*)alloc((size_t)NTOK * H * 4);
  float* af   = (float*)alloc((size_t)NTOK * H * 4);
  float* tmpf = (float*)alloc((size_t)NTOK * H * 4);
  unsigned short* xb   = (unsigned short*)alloc((size_t)NTOK * H * 2);
  unsigned short* ab   = (unsigned short*)alloc((size_t)NTOK * H * 2);
  unsigned short* qbuf = (unsigned short*)alloc((size_t)NTOK * H * 2);
  unsigned short* kbuf = (unsigned short*)alloc((size_t)NTOK * H * 2);  // contiguous after qbuf
  unsigned short* vbuf = (unsigned short*)alloc((size_t)NTOK * H * 2);
  unsigned short* cbuf = (unsigned short*)alloc((size_t)NTOK * H * 2);
  unsigned short* hbuf = qbuf;  // FFN hidden [NTOK, IDIM] aliases q+k (dead by then)

  unsigned short* wq = (unsigned short*)alloc((size_t)LAYERS * H * H * 2);
  unsigned short* wk = (unsigned short*)alloc((size_t)LAYERS * H * H * 2);
  unsigned short* wv = (unsigned short*)alloc((size_t)LAYERS * H * H * 2);
  unsigned short* wo = (unsigned short*)alloc((size_t)LAYERS * H * H * 2);
  unsigned short* wi = (unsigned short*)alloc((size_t)LAYERS * H * IDIM * 2);
  unsigned short* w2 = (unsigned short*)alloc((size_t)LAYERS * IDIM * H * 2);
  unsigned short* wp = (unsigned short*)alloc((size_t)H * H * 2);

  // weight f32 -> bf16 transposed conversion (tiny; deterministic per launch)
  {
    const int nhh = LAYERS * H * H, nhi = LAYERS * H * IDIM, npp = H * H;
    convert_T_kernel<<<(nhh + 255) / 256, 256, 0, stream>>>(qw, wq, H, H, nhh);
    convert_T_kernel<<<(nhh + 255) / 256, 256, 0, stream>>>(kw, wk, H, H, nhh);
    convert_T_kernel<<<(nhh + 255) / 256, 256, 0, stream>>>(vw, wv, H, H, nhh);
    convert_T_kernel<<<(nhh + 255) / 256, 256, 0, stream>>>(ow, wo, H, H, nhh);
    convert_T_kernel<<<(nhi + 255) / 256, 256, 0, stream>>>(iw, wi, H, IDIM, nhi);
    convert_T_kernel<<<(nhi + 255) / 256, 256, 0, stream>>>(outw, w2, IDIM, H, nhi);
    convert_T_kernel<<<(npp + 255) / 256, 256, 0, stream>>>(pool_w, wp, H, H, npp);
  }

  embed_ln_kernel<<<NTOK, 256, 0, stream>>>(word, age, seg, posi,
                                            word_emb, seg_emb, age_emb, posi_emb,
                                            emb_ln_g, emb_ln_b, xf, xb);

  const dim3 gH(H / 128, NTOK / 64);      // (2, 1600)
  const dim3 gI(IDIM / 128, NTOK / 64);   // (4, 1600)
  const dim3 gA(NSEQ, NHEADS);            // (1600, 4)

  for (int l = 0; l < LAYERS; ++l) {
    gemm_wmma<EPI_BF16><<<gH, 256, 0, stream>>>(xb, H, wq + (size_t)l * H * H,
                                                qb + l * H, nullptr, nullptr, qbuf, H);
    gemm_wmma<EPI_BF16><<<gH, 256, 0, stream>>>(xb, H, wk + (size_t)l * H * H,
                                                kb + l * H, nullptr, nullptr, kbuf, H);
    gemm_wmma<EPI_BF16><<<gH, 256, 0, stream>>>(xb, H, wv + (size_t)l * H * H,
                                                vb + l * H, nullptr, nullptr, vbuf, H);

    attn_kernel<<<gA, 256, 0, stream>>>(qbuf, kbuf, vbuf, mask, cbuf);

    gemm_wmma<EPI_RES_F32><<<gH, 256, 0, stream>>>(cbuf, H, wo + (size_t)l * H * H,
                                                   ob + l * H, xf, tmpf, nullptr, H);
    layernorm_kernel<<<NTOK, 256, 0, stream>>>(tmpf, attn_ln_g + l * H, attn_ln_b + l * H,
                                               af, ab);

    gemm_wmma<EPI_GELU_BF16><<<gI, 256, 0, stream>>>(ab, H, wi + (size_t)l * H * IDIM,
                                                     ib + l * IDIM, nullptr, nullptr, hbuf, H);
    gemm_wmma<EPI_RES_F32><<<gH, 256, 0, stream>>>(hbuf, IDIM, w2 + (size_t)l * IDIM * H,
                                                   ffn_ob + l * H, af, tmpf, nullptr, IDIM);
    layernorm_kernel<<<NTOK, 256, 0, stream>>>(tmpf, out_ln_g + l * H, out_ln_b + l * H,
                                               xf, xb);
  }

  // pooler: rows = first token of each sequence (lda = SEQ*H), tanh epilogue -> d_out
  const dim3 gP(H / 128, NSEQ / 64);      // (2, 25)
  gemm_wmma<EPI_TANH_F32><<<gP, 256, 0, stream>>>(xb, SEQ * H, wp,
                                                  pool_b, nullptr, out, nullptr, H);
}
